// Attention_31842887532911
// MI455X (gfx1250) — matrix operand order repaired and checked by reading, not compiled, not run on hardware
//
#include <hip/hip_runtime.h>
#include <hip/hip_bf16.h>

// ---------------- problem constants ----------------
#define B_   2
#define T_   2048
#define D_   2048
#define N_   8
#define H_   256
#define L_   16
#define M_   (B_*T_)        // 4096 rows of "tokens"
#define NH_  (N_*H_)        // 2048
#define NL_  (N_*L_)        // 128
#define LP_  32             // LoRA rank padded to one WMMA K-step
#define BIG_NEG (-2.3819763e+38f)

typedef __bf16 bh;
typedef __attribute__((ext_vector_type(16))) __bf16 v16bf;
typedef __attribute__((ext_vector_type(8)))  __bf16 v8bf;
typedef __attribute__((ext_vector_type(8)))  float  v8f;

// ---------------- WMMA wrapper ----------------
__device__ __forceinline__ v8f wmma_bf16(v16bf a, v16bf b, v8f c) {
#if defined(__HIP_DEVICE_COMPILE__)
    return __builtin_amdgcn_wmma_f32_16x16x32_bf16(
        /*neg_a=*/false, a, /*neg_b=*/false, b,
        /*c_mod=*/(short)0, c, /*reuse_a=*/false, /*reuse_b=*/false);
#else
    (void)a; (void)b; return c;
#endif
}

// A-fragment (16x32 bf16, row-major source, row = lane&15):
//   lanes 0-15 : K = k0+0..15
//   lanes 16-31: K = k0+16..31
__device__ __forceinline__ v16bf combine8(v8bf lo, v8bf hi8) {
    v16bf r;
#pragma unroll
    for (int i = 0; i < 8; ++i) { r[i] = lo[i]; r[i + 8] = hi8[i]; }
    return r;
}
__device__ __forceinline__ v16bf load_a_frag(const bh* base, int ld, int k0, int lane) {
    const int m  = lane & 15;
    const int hi = lane >> 4;
    const bh* p = base + (size_t)m * ld + k0 + hi * 16;
    return combine8(*(const v8bf*)p, *(const v8bf*)(p + 8));
}

// cross-16-lane row reductions for the C-fragment layout
__device__ __forceinline__ float rmax16(float v) {
#pragma unroll
    for (int o = 8; o >= 1; o >>= 1) v = fmaxf(v, __shfl_xor(v, o, 16));
    return v;
}
__device__ __forceinline__ float rsum16(float v) {
#pragma unroll
    for (int o = 8; o >= 1; o >>= 1) v += __shfl_xor(v, o, 16);
    return v;
}

// ---------------- conversion kernels ----------------
// f32 [rows x C] -> bf16 [rows x Cp], zero pad columns C..Cp
__global__ void cvt_pad_kernel(const float* __restrict__ src, bh* __restrict__ dst,
                               int rows, int C, int Cp) {
    long i = (long)blockIdx.x * blockDim.x + threadIdx.x;
    if (i >= (long)rows * Cp) return;
    int r = (int)(i / Cp), c = (int)(i % Cp);
    dst[i] = (c < C) ? (bh)src[(long)r * C + c] : (bh)0.0f;
}

// batched transpose: src f32 [batch][R][C] -> dst bf16 [batch][C][Rp] (pad R..Rp with 0)
__global__ void transpose_cvt_kernel(const float* __restrict__ src, bh* __restrict__ dst,
                                     int R, int C, int Rp, long sStride, long dStride) {
    long i = (long)blockIdx.x * blockDim.x + threadIdx.x;
    if (i >= (long)C * Rp) return;
    int c = (int)(i / Rp), r = (int)(i % Rp);
    int b = blockIdx.z;
    dst[(long)b * dStride + i] =
        (r < R) ? (bh)src[(long)b * sStride + (long)r * C + c] : (bh)0.0f;
}

// ---------------- generic bf16 WMMA GEMM ----------------
// C[M x Ncols] (f32) (beta? += : =) A[M x K](bf16,row-major) * Bt[Ncols x K](bf16)^T
// grid: (Ncols/(16*JT), M/(128*MT)), block 256 = 8 waves; wave w owns a (16*MT) x (16*JT)
// tile at m0 = (by*8+w)*16*MT, n0 = bx*16*JT (scalar -> no divergent branches, EXEC all-1).
// MT*JT accumulator tiles give A/B fragment reuse and 8 independent WMMA chains.
template <int MT, int JT>
__global__ __launch_bounds__(256) void gemm_bf16_kernel(
    const bh* __restrict__ A, const bh* __restrict__ Bt, float* __restrict__ C,
    int Ncols, int Kdim, int beta) {
    const int lane = threadIdx.x & 31;
    const int wv   = threadIdx.x >> 5;
    const int m0   = (blockIdx.y * 8 + wv) * (16 * MT);
    const int n0   = blockIdx.x * (16 * JT);
    const int li   = lane & 15, hi = lane >> 4;

    const bh* aRow[MT];
#pragma unroll
    for (int i = 0; i < MT; ++i)
        aRow[i] = A + (size_t)(m0 + i * 16 + li) * Kdim + hi * 16;  // +k0 / +k0+8 chunks
    const bh* bRow[JT];
#pragma unroll
    for (int j = 0; j < JT; ++j)
        bRow[j] = Bt + (size_t)(n0 + j * 16 + li) * Kdim + hi * 16;

    v8f acc[MT][JT];
#pragma unroll
    for (int i = 0; i < MT; ++i)
#pragma unroll
        for (int j = 0; j < JT; ++j)
#pragma unroll
            for (int r = 0; r < 8; ++r) acc[i][j][r] = 0.0f;

    // software pipeline: fetch k0+32 while multiplying k0
    v16bf a[MT], b[JT];
#pragma unroll
    for (int i = 0; i < MT; ++i)
        a[i] = combine8(*(const v8bf*)(aRow[i]), *(const v8bf*)(aRow[i] + 8));
#pragma unroll
    for (int j = 0; j < JT; ++j) b[j] = *(const v16bf*)(bRow[j]);

    for (int k0 = 32; k0 < Kdim; k0 += 32) {
        v16bf an[MT], bn[JT];
#pragma unroll
        for (int i = 0; i < MT; ++i)
            an[i] = combine8(*(const v8bf*)(aRow[i] + k0), *(const v8bf*)(aRow[i] + k0 + 8));
#pragma unroll
        for (int j = 0; j < JT; ++j) bn[j] = *(const v16bf*)(bRow[j] + k0);
#pragma unroll
        for (int i = 0; i < MT; ++i)
#pragma unroll
            for (int j = 0; j < JT; ++j) acc[i][j] = wmma_bf16(a[i], b[j], acc[i][j]);
#pragma unroll
        for (int i = 0; i < MT; ++i) a[i] = an[i];
#pragma unroll
        for (int j = 0; j < JT; ++j) b[j] = bn[j];
    }
#pragma unroll
    for (int i = 0; i < MT; ++i)
#pragma unroll
        for (int j = 0; j < JT; ++j) acc[i][j] = wmma_bf16(a[i], b[j], acc[i][j]);

#pragma unroll
    for (int i = 0; i < MT; ++i)
#pragma unroll
        for (int j = 0; j < JT; ++j)
#pragma unroll
            for (int r = 0; r < 8; ++r) {
                const size_t idx =
                    (size_t)(m0 + i * 16 + r + hi * 8) * Ncols + n0 + j * 16 + li;
                float v = acc[i][j][r];
                if (beta) v += C[idx];
                C[idx] = v;
            }
}

// ---------------- RoPE / repack kernels ----------------
// Q f32 [B,T,N,H] -> rope -> *H^-0.5 -> bf16 [B,N,T,H]
__global__ void rope_q_kernel(const float* __restrict__ Qf, const int* __restrict__ pos,
                              bh* __restrict__ Qb) {
    int idx = blockIdx.x * blockDim.x + threadIdx.x;
    if (idx >= B_ * T_ * N_ * (H_ / 2)) return;
    int hh = idx & 127;
    int n  = (idx >> 7) & (N_ - 1);
    int t  = (idx >> 10) & (T_ - 1);
    int b  = idx >> 21;
    const float* q = Qf + (size_t)(b * T_ + t) * NH_ + n * H_;
    float x1 = q[hh], x2 = q[hh + 128];
    float p = (float)pos[b * T_ + t];
    float inv_ts = __expf(-(float)hh * (9.210340371976184f / 128.0f)); // 10000^(-hh/128)
    float rad = p * inv_ts, s, c;
    __sincosf(rad, &s, &c);
    bh* o = Qb + (((size_t)b * N_ + n) * T_ + t) * H_;
    o[hh]       = (bh)((x1 * c - x2 * s) * 0.0625f);
    o[hh + 128] = (bh)((x2 * c + x1 * s) * 0.0625f);
}

// K f32 [B,T,H] -> rope -> bf16 [B,T,H]
__global__ void rope_k_kernel(const float* __restrict__ Kf, const int* __restrict__ pos,
                              bh* __restrict__ Kb) {
    int idx = blockIdx.x * blockDim.x + threadIdx.x;
    if (idx >= B_ * T_ * (H_ / 2)) return;
    int hh = idx & 127;
    int t  = (idx >> 7) & (T_ - 1);
    int b  = idx >> 18;
    const float* k = Kf + (size_t)(b * T_ + t) * H_;
    float x1 = k[hh], x2 = k[hh + 128];
    float p = (float)pos[b * T_ + t];
    float inv_ts = __expf(-(float)hh * (9.210340371976184f / 128.0f));
    float rad = p * inv_ts, s, c;
    __sincosf(rad, &s, &c);
    bh* o = Kb + (size_t)(b * T_ + t) * H_;
    o[hh]       = (bh)(x1 * c - x2 * s);
    o[hh + 128] = (bh)(x2 * c + x1 * s);
}

// V f32 [B,T,H] -> bf16 [B,H,T] (transposed so PV B-fragments are contiguous)
__global__ void cvt_v_t_kernel(const float* __restrict__ Vf, bh* __restrict__ Vt) {
    int idx = blockIdx.x * blockDim.x + threadIdx.x;
    if (idx >= B_ * T_ * H_) return;
    int h = idx & (H_ - 1);
    int t = (idx >> 8) & (T_ - 1);
    int b = idx >> 19;
    Vt[((size_t)b * H_ + h) * T_ + t] = (bh)Vf[idx];
}

// ---------------- flash attention (causal, GQA K=1) ----------------
// grid (T/16, N, B), block 32 (1 wave). Q [B,N,T,H], K [B,T,H], Vt [B,H,T] -> enc bf16 [B*T, N*H]
// Q tile staged through LDS; compiler keeps the 8 A-fragments resident in VGPRs across
// the key loop (verified in round-3 asm: pipelined clauses + partial loadcnt waits).
__global__ __launch_bounds__(32) void attn_kernel(
    const bh* __restrict__ Qb, const bh* __restrict__ Kb,
    const bh* __restrict__ Vt, bh* __restrict__ Encb) {
    __shared__ bh Qs[16 * H_];   // 8 KB: Q tile (16 rows x 256, row-major)
    __shared__ bh Pl[16 * 32];   // 1 KB: P re-layout buffer
    const int lane = threadIdx.x;
    const int li = lane & 15, hi = lane >> 4;
    const int t0 = blockIdx.x * 16;
    const int n  = blockIdx.y;
    const int b  = blockIdx.z;

    const bh* Qp = Qb + (((size_t)b * N_ + n) * T_ + t0) * H_;  // 16 contiguous rows
    const bh* Kp = Kb + (size_t)b * T_ * H_;
    const bh* Vp = Vt + (size_t)b * H_ * T_;

    // cooperative copy of the contiguous 8KB Q tile into LDS (b128 per lane)
#pragma unroll
    for (int i = 0; i < 16; ++i) {
        const int off = (i * 32 + lane) * 8;   // 8 bf16 = 16 bytes per lane
        *(v8bf*)(Qs + off) = *(const v8bf*)(Qp + off);
    }

    v8f acc[16];
#pragma unroll
    for (int j = 0; j < 16; ++j)
#pragma unroll
        for (int r = 0; r < 8; ++r) acc[j][r] = 0.0f;

    float mrow[8], lsum[8];
#pragma unroll
    for (int r = 0; r < 8; ++r) { mrow[r] = BIG_NEG; lsum[r] = 0.0f; }

    for (int s0 = 0; s0 < t0 + 16; s0 += 32) {   // wave-uniform causal bound
        const bh* ka = Kp + (size_t)(s0 + li) * H_ + hi * 16;
        const bh* kb = Kp + (size_t)(s0 + 16 + li) * H_ + hi * 16;
        v8f Sa, Sb;
#pragma unroll
        for (int r = 0; r < 8; ++r) { Sa[r] = 0.0f; Sb[r] = 0.0f; }
        // pipelined QK^T: global K fragments fetched one kc ahead; A-fragments from LDS
        v16bf ba = *(const v16bf*)(ka);
        v16bf bb = *(const v16bf*)(kb);
#pragma unroll
        for (int kc = 0; kc < 8; ++kc) {
            v16bf ban = ba, bbn = bb;
            if (kc < 7) {
                ban = *(const v16bf*)(ka + (kc + 1) * 32);
                bbn = *(const v16bf*)(kb + (kc + 1) * 32);
            }
            v16bf aq = load_a_frag(Qs, H_, kc * 32, lane);   // ds_load_b128 x2
            Sa = wmma_bf16(aq, ba, Sa);
            Sb = wmma_bf16(aq, bb, Sb);
            ba = ban; bb = bbn;
        }
        // online softmax in C-fragment layout: lane holds col li, rows r+hi*8
#pragma unroll
        for (int r = 0; r < 8; ++r) {
            const int row = t0 + r + hi * 8;
            float sa = (s0 + li      <= row) ? Sa[r] : BIG_NEG;
            float sb = (s0 + 16 + li <= row) ? Sb[r] : BIG_NEG;
            float mnew = fmaxf(mrow[r], rmax16(fmaxf(sa, sb)));
            float corr = __expf(mrow[r] - mnew);
            float pa = __expf(sa - mnew);
            float pb = __expf(sb - mnew);
            lsum[r] = lsum[r] * corr + rsum16(pa + pb);
            mrow[r] = mnew;
#pragma unroll
            for (int j = 0; j < 16; ++j) acc[j][r] *= corr;
            Pl[(r + hi * 8) * 32 + li]      = (bh)pa;   // re-layout P via LDS
            Pl[(r + hi * 8) * 32 + 16 + li] = (bh)pb;
        }
        // single wave: LDS pipeline is in-order; compiler inserts s_wait_dscnt before use
        v16bf pf = load_a_frag(Pl, 32, 0, lane);
        // pipelined PV: fetch j+1 while multiplying j
        v16bf vb = *(const v16bf*)(Vp + (size_t)li * T_ + s0 + hi * 16);
#pragma unroll
        for (int j = 0; j < 16; ++j) {
            v16bf vbn = vb;
            if (j < 15)
                vbn = *(const v16bf*)(Vp + (size_t)((j + 1) * 16 + li) * T_ + s0 + hi * 16);
            acc[j] = wmma_bf16(pf, vb, acc[j]);
            vb = vbn;
        }
    }
    // epilogue: normalize and scatter enc rows (bf16, [B*T, N*H])
#pragma unroll
    for (int r = 0; r < 8; ++r) {
        const int row = t0 + r + hi * 8;
        const float inv = 1.0f / lsum[r];
        bh* op = Encb + ((size_t)b * T_ + row) * NH_ + (size_t)n * H_;
#pragma unroll
        for (int j = 0; j < 16; ++j) op[j * 16 + li] = (bh)(acc[j][r] * inv);
    }
}

// ---------------- host side ----------------
static inline void launch_gemm(const bh* A, const bh* Bt, float* C,
                               int M, int Ncols, int Kdim, int beta, hipStream_t s) {
    if ((Ncols & 63) == 0) {
        dim3 g((unsigned)(Ncols / 64), (unsigned)(M / 256));
        gemm_bf16_kernel<2, 4><<<g, 256, 0, s>>>(A, Bt, C, Ncols, Kdim, beta);
    } else {  // Ncols multiple of 16 (LoRA rank GEMMs, Ncols=16)
        dim3 g((unsigned)(Ncols / 16), (unsigned)(M / 256));
        gemm_bf16_kernel<2, 1><<<g, 256, 0, s>>>(A, Bt, C, Ncols, Kdim, beta);
    }
}
static inline void launch_tcvt(const float* src, bh* dst, int batch, int R, int C, int Rp,
                               long sStride, hipStream_t s) {
    long elems = (long)C * Rp;
    dim3 g((unsigned)((elems + 255) / 256), 1, (unsigned)batch);
    transpose_cvt_kernel<<<g, 256, 0, s>>>(src, dst, R, C, Rp, sStride, (long)C * Rp);
}
static inline void launch_cvt(const float* src, bh* dst, int rows, int C, int Cp, hipStream_t s) {
    long elems = (long)rows * Cp;
    cvt_pad_kernel<<<(unsigned)((elems + 255) / 256), 256, 0, s>>>(src, dst, rows, C, Cp);
}

extern "C" void kernel_launch(void* const* d_in, const int* in_sizes, int n_in,
                              void* d_out, int out_size, void* d_ws, size_t ws_size,
                              hipStream_t stream) {
    const float* x          = (const float*)d_in[0];
    const int*   positions  = (const int*)d_in[1];
    // d_in[2] attn_mask (causal, regenerated analytically), d_in[3] decode==0: unused
    const float* q_w        = (const float*)d_in[4];
    const float* q_lora_a   = (const float*)d_in[5];
    const float* q_lora_b   = (const float*)d_in[6];
    const float* kv_w       = (const float*)d_in[7];
    const float* kv_lora_a  = (const float*)d_in[8];
    const float* kv_lora_b  = (const float*)d_in[9];
    const float* out_w      = (const float*)d_in[10];
    const float* out_lora_a = (const float*)d_in[11];
    const float* out_lora_b = (const float*)d_in[12];
    float* out = (float*)d_out;

    char* ws = (char*)d_ws;
    size_t off = 0;
    auto carve = [&](size_t bytes) -> char* {
        char* p = ws + off;
        off = (off + bytes + 255) & ~(size_t)255;
        return p;
    };
    bh*    Xb    = (bh*)carve((size_t)M_ * D_ * 2);
    bh*    Wq_t  = (bh*)carve((size_t)NH_ * D_ * 2);
    bh*    Wqa_t = (bh*)carve((size_t)NL_ * D_ * 2);
    bh*    Wqb_t = (bh*)carve((size_t)NH_ * NL_ * 2);
    bh*    Wkv_t = (bh*)carve((size_t)2 * H_ * D_ * 2);
    bh*    Wkva_t= (bh*)carve((size_t)2 * L_ * D_ * 2);
    bh*    Wkvb_t= (bh*)carve((size_t)2 * H_ * LP_ * 2);
    bh*    Wo_t  = (bh*)carve((size_t)D_ * NH_ * 2);
    bh*    Woa_t = (bh*)carve((size_t)NL_ * NH_ * 2);
    bh*    Wob_t = (bh*)carve((size_t)D_ * NL_ * 2);
    float* Qf    = (float*)carve((size_t)M_ * NH_ * 4);
    float* Kf    = (float*)carve((size_t)M_ * H_ * 4);
    float* Vf    = (float*)carve((size_t)M_ * H_ * 4);
    float* QAf   = (float*)carve((size_t)M_ * NL_ * 4);
    bh*    QAb   = (bh*)carve((size_t)M_ * NL_ * 2);
    float* KAf   = (float*)carve((size_t)M_ * L_ * 4);
    float* VAf   = (float*)carve((size_t)M_ * L_ * 4);
    bh*    KAb   = (bh*)carve((size_t)M_ * LP_ * 2);
    bh*    VAb   = (bh*)carve((size_t)M_ * LP_ * 2);
    bh*    Qbq   = (bh*)carve((size_t)M_ * NH_ * 2);
    bh*    Kbb   = (bh*)carve((size_t)M_ * H_ * 2);
    bh*    Vtt   = (bh*)carve((size_t)M_ * H_ * 2);
    bh*    Encb  = (bh*)carve((size_t)M_ * NH_ * 2);
    float* OAf   = (float*)carve((size_t)M_ * NL_ * 4);
    bh*    OAb   = (bh*)carve((size_t)M_ * NL_ * 2);
    (void)ws_size; (void)in_sizes; (void)n_in; (void)out_size;

    // ---- stage 0: convert x, pack weights to bf16 B^T layouts ----
    launch_cvt(x, Xb, M_, D_, D_, stream);
    launch_tcvt(q_w,        Wq_t,  N_, D_,  H_,  D_,  (long)D_ * H_,  stream); // [N*H][D]
    launch_tcvt(q_lora_a,   Wqa_t, N_, D_,  L_,  D_,  (long)D_ * L_,  stream); // [N*L][D]
    launch_tcvt(q_lora_b,   Wqb_t, 1,  NL_, NH_, NL_, (long)NL_ * NH_, stream); // [N*H][N*L]
    launch_tcvt(kv_w,       Wkv_t, 2,  D_,  H_,  D_,  (long)D_ * H_,  stream); // [2][H][D]
    launch_tcvt(kv_lora_a,  Wkva_t,2,  D_,  L_,  D_,  (long)D_ * L_,  stream); // [2][L][D]
    launch_tcvt(kv_lora_b,  Wkvb_t,2,  L_,  H_,  LP_, (long)L_ * H_,  stream); // [2][H][32] padded
    launch_tcvt(out_w,      Wo_t,  1,  NH_, D_,  NH_, (long)NH_ * D_, stream); // [D][N*H]
    launch_tcvt(out_lora_a, Woa_t, 1,  NH_, NL_, NH_, (long)NH_ * NL_, stream); // [N*L][N*H]
    launch_tcvt(out_lora_b, Wob_t, 1,  NL_, D_,  NL_, (long)NL_ * D_, stream); // [D][N*L]

    // ---- stage 1: projections (+LoRA accumulate, scale==1) ----
    launch_gemm(Xb, Wq_t,  Qf,  M_, NH_, D_, 0, stream);
    launch_gemm(Xb, Wqa_t, QAf, M_, NL_, D_, 0, stream);
    launch_cvt(QAf, QAb, M_, NL_, NL_, stream);
    launch_gemm(QAb, Wqb_t, Qf, M_, NH_, NL_, 1, stream);

    launch_gemm(Xb, Wkv_t,             Kf, M_, H_, D_, 0, stream);
    launch_gemm(Xb, Wkv_t + (size_t)H_ * D_, Vf, M_, H_, D_, 0, stream);
    launch_gemm(Xb, Wkva_t,             KAf, M_, L_, D_, 0, stream);
    launch_gemm(Xb, Wkva_t + (size_t)L_ * D_, VAf, M_, L_, D_, 0, stream);
    launch_cvt(KAf, KAb, M_, L_, LP_, stream);
    launch_cvt(VAf, VAb, M_, L_, LP_, stream);
    launch_gemm(KAb, Wkvb_t,               Kf, M_, H_, LP_, 1, stream);
    launch_gemm(VAb, Wkvb_t + (size_t)H_ * LP_, Vf, M_, H_, LP_, 1, stream);

    // ---- stage 2: RoPE + repack to attention layouts ----
    rope_q_kernel<<<(B_ * T_ * N_ * (H_ / 2)) / 256, 256, 0, stream>>>(Qf, positions, Qbq);
    rope_k_kernel<<<(B_ * T_ * (H_ / 2)) / 256, 256, 0, stream>>>(Kf, positions, Kbb);
    cvt_v_t_kernel<<<(B_ * T_ * H_) / 256, 256, 0, stream>>>(Vf, Vtt);

    // ---- stage 3: causal flash attention ----
    {
        dim3 g(T_ / 16, N_, B_);
        attn_kernel<<<g, 32, 0, stream>>>(Qbq, Kbb, Vtt, Encb);
    }

    // ---- stage 4: output projection (+LoRA) into d_out ----
    launch_gemm(Encb, Wo_t,  out, M_, D_,  NH_, 0, stream);
    launch_gemm(Encb, Woa_t, OAf, M_, NL_, NH_, 0, stream);
    launch_cvt(OAf, OAb, M_, NL_, NL_, stream);
    launch_gemm(OAb, Wob_t, out, M_, D_, NL_, 1, stream);
}
